// SBShortKeyMemory_28587302323146
// MI455X (gfx1250) — compile-verified, matrix operand order checked by reading, not run on hardware
//
#include <hip/hip_runtime.h>
#include <math.h>

#define KB constexpr int
KB B_ = 2048, D_ = 512, N_ = 64, M_ = 32;

typedef __attribute__((ext_vector_type(16))) __bf16 v16bf;
typedef __attribute__((ext_vector_type(8)))  float  v8f;

__device__ __forceinline__ float wave_sum(float v) {
    v += __shfl_xor(v, 16);
    v += __shfl_xor(v, 8);
    v += __shfl_xor(v, 4);
    v += __shfl_xor(v, 2);
    v += __shfl_xor(v, 1);
    return v;
}
__device__ __forceinline__ float dot4(float4 a, float4 b) {
    return a.x*b.x + a.y*b.y + a.z*b.z + a.w*b.w;
}
__device__ __forceinline__ float sigm(float x) { return 1.0f / (1.0f + expf(-x)); }
__device__ __forceinline__ float clamp01(float x) { return fminf(fmaxf(x, 0.0f), 1.0f); }

__device__ __forceinline__ v16bf pack16(float4 a, float4 b, float4 c, float4 d) {
    v16bf r;
    r[0]=(__bf16)a.x; r[1]=(__bf16)a.y; r[2]=(__bf16)a.z; r[3]=(__bf16)a.w;
    r[4]=(__bf16)b.x; r[5]=(__bf16)b.y; r[6]=(__bf16)b.z; r[7]=(__bf16)b.w;
    r[8]=(__bf16)c.x; r[9]=(__bf16)c.y; r[10]=(__bf16)c.z; r[11]=(__bf16)c.w;
    r[12]=(__bf16)d.x; r[13]=(__bf16)d.y; r[14]=(__bf16)d.z; r[15]=(__bf16)d.w;
    return r;
}

// ---------------- Kernel 1: q·k dot and key inverse norms, one wave per (b,n) --
__global__ __launch_bounds__(256) void k_dotnorm(
    const float* __restrict__ q, const float* __restrict__ keys,
    float* __restrict__ qdot, float* __restrict__ inv_norm) {
    const int wave = threadIdx.x >> 5, lane = threadIdx.x & 31;
    const int bn = blockIdx.x * 8 + wave;
    const int b = bn >> 6;                 // N_ = 64
    const float4* kp = (const float4*)(keys + (size_t)bn * D_) + lane * 4;
    const float4* qp = (const float4*)(q + (size_t)b * D_) + lane * 4;
    float dot = 0.f, ss = 0.f;
#pragma unroll
    for (int j = 0; j < 4; ++j) {
        float4 k4 = kp[j], q4 = qp[j];
        ss  += dot4(k4, k4);
        dot += dot4(k4, q4);
    }
    dot = wave_sum(dot);
    ss  = wave_sum(ss);
    if (lane == 0) {
        qdot[bn] = dot;
        inv_norm[bn] = 1.0f / fmaxf(sqrtf(ss), 1e-6f);
    }
}

// ---------------- Kernel 2: per-batch episodic argmax, scores/top2/read_out, gates
__global__ __launch_bounds__(256) void k_perbatch(
    const float* __restrict__ qdot, const float* __restrict__ inv_norm,
    const float* __restrict__ delay, const float* __restrict__ sal,
    const float* __restrict__ s_strength, const float* __restrict__ s_age,
    const float* __restrict__ s_usage, const float* __restrict__ s_values,
    const float* __restrict__ signal, const float* __restrict__ hidden,
    const float* __restrict__ branch, const float* __restrict__ entropy,
    const float* __restrict__ ep_vals, const float* __restrict__ ep_strength,
    const float* __restrict__ ep_hits, const float* __restrict__ ep_age,
    const float* __restrict__ fg_W, const float* __restrict__ fg_b,
    const float* __restrict__ pg_W, const float* __restrict__ pg_b,
    const float* __restrict__ cg_W, const float* __restrict__ cg_b,
    float* __restrict__ scal, float* __restrict__ out_read, float* __restrict__ out_hits) {
    const int b = blockIdx.x, t = threadIdx.x;
    const int wave = t >> 5, lane = t & 31;
    __shared__ float s_pri[M_];
    __shared__ float s_sc[N_];
    __shared__ float red0[256], red1[256], red2[256];
    __shared__ float s_w[2];
    __shared__ int   s_i[2];
    __shared__ float s_conf;

    // episodic value norms -> priority
#pragma unroll
    for (int i = 0; i < 4; ++i) {
        const int m = wave + 8 * i;
        const float4* p = (const float4*)(ep_vals + ((size_t)b * M_ + m) * D_) + lane * 4;
        float ss = 0.f;
#pragma unroll
        for (int j = 0; j < 4; ++j) { float4 x = p[j]; ss += dot4(x, x); }
        ss = wave_sum(ss);
        if (lane == 0) {
            float en = sqrtf(ss) * 0.044194173824159216f;   // / sqrt(512)
            s_pri[m] = 0.45f * ep_strength[b*M_+m] + 0.3f * (ep_hits[b*M_+m] * (1.f/6.f))
                     + 0.15f * (1.f - ep_age[b*M_+m]) + 0.1f * clamp01(en);
        }
    }
    __syncthreads();
    if (t == 0) {
        int si = 0; float best = s_pri[0];
        for (int m = 1; m < M_; ++m) if (s_pri[m] > best) { best = s_pri[m]; si = m; }
        scal[b*16 + 0] = __int_as_float(si);
        scal[b*16 + 1] = best;
        s_conf = best;
    }
    // combined scores
    if (t < N_) {
        const int n = t;
        float ks = qdot[b*N_+n] * inv_norm[b*N_+n];
        float st = s_strength[b*N_+n], ag = s_age[b*N_+n], us = s_usage[b*N_+n];
        float pers = 2.8f*st + 0.9f*us + 0.6f*(1.f - ag);
        s_sc[n] = 0.65f*ks + 0.22f*pers + 0.08f*delay[b]*ag + 0.05f*sal[b]*st;
    }
    __syncthreads();
    if (t == 0) {
        int i0 = 0; float v0 = s_sc[0];
        for (int n = 1; n < N_; ++n) if (s_sc[n] > v0) { v0 = s_sc[n]; i0 = n; }
        int i1 = (i0 == 0) ? 1 : 0; float v1 = s_sc[i1];
        for (int n = 0; n < N_; ++n) if (n != i0 && s_sc[n] > v1) { v1 = s_sc[n]; i1 = n; }
        float e1 = expf((v1 - v0) * 2.0f);      // /TEMP = *2
        float w0 = 1.f / (1.f + e1);
        s_i[0] = i0; s_i[1] = i1; s_w[0] = w0; s_w[1] = e1 * w0;
    }
    __syncthreads();
    const int i0 = s_i[0], i1 = s_i[1];
    const float w0 = s_w[0], w1 = s_w[1];
    if (t < 128) {  // 128 x float4 = 512 floats
        const float4 a = ((const float4*)(s_values + ((size_t)b*N_ + i0)*D_))[t];
        const float4 c = ((const float4*)(s_values + ((size_t)b*N_ + i1)*D_))[t];
        float4 r;
        r.x = w0*a.x + w1*c.x; r.y = w0*a.y + w1*c.y;
        r.z = w0*a.z + w1*c.z; r.w = w0*a.w + w1*c.w;
        ((float4*)(out_read + (size_t)b*D_))[t] = r;
    }
    if (t < N_) {
        float h = 0.f;
        if (t == i0) h += w0;
        if (t == i1) h += w1;
        out_hits[b*N_ + t] = h;
    }
    // gate dot products over routed (len 2D+3)
    float af = 0.f, ap = 0.f, ac = 0.f;
    for (int j = t; j < 2*D_ + 3; j += 256) {
        float rj = (j < D_) ? signal[(size_t)b*D_ + j]
                 : (j < 2*D_) ? hidden[(size_t)b*D_ + j - D_]
                 : branch[b*3 + (j - 2*D_)];
        af += rj * fg_W[j]; ap += rj * pg_W[j]; ac += rj * cg_W[j];
    }
    red0[t] = af; red1[t] = ap; red2[t] = ac;
    __syncthreads();
    for (int s = 128; s > 0; s >>= 1) {
        if (t < s) { red0[t] += red0[t+s]; red1[t] += red1[t+s]; red2[t] += red2[t+s]; }
        __syncthreads();
    }
    if (t == 0) {
        float fb   = sigm(red0[0] + fg_b[0]);
        float pg   = sigm(red1[0] + pg_b[0]);
        float cons = sigm(red2[0] + cg_b[0] + 2.2f * (s_conf - 0.5f));
        float comp = sigm((0.72f - entropy[b]) * 5.5f);
        float kf   = clamp01(0.45f*fb + 0.3f*comp + 0.25f*delay[b]);
        scal[b*16+2] = cons; scal[b*16+3] = pg; scal[b*16+4] = kf; scal[b*16+5] = comp;
    }
}

// ---------------- Kernel 3: bf16 WMMA GEMM (base_key & base_value) + mix epilogue
__global__ __launch_bounds__(256) void k_gemm_mix(
    const float* __restrict__ signal, const float* __restrict__ hidden,
    const float* __restrict__ key_W, const float* __restrict__ key_b,
    const float* __restrict__ val_W, const float* __restrict__ val_b,
    const float* __restrict__ ep_keys, const float* __restrict__ ep_vals,
    const float* __restrict__ scal,
    float* __restrict__ cand_key, float* __restrict__ cand_val) {
    const int wave = threadIdx.x >> 5, lane = threadIdx.x & 31;
    const int w  = blockIdx.x * 8 + wave;     // 4096 waves total
    const int tm = w >> 5;                    // 128 M-tiles (batch)
    const int tn = w & 31;                    // 32 N-tiles (features)
    const int mrow = tm * 16 + (lane & 15);   // A row (batch)
    const int ncol = tn * 16 + (lane & 15);   // B col (feature d)
    const int kb   = (lane >> 4) * 8;         // K-chunk base within 16

    v8f acck = {}; v8f accv = {};
    for (int k0 = 0; k0 < 2 * D_; k0 += 32) {
        const int ka = k0 + kb;
        const int kc = k0 + 16 + kb;
        // A = joined[b, k]  (signal | hidden), contiguous 8-float chunks
        const float* ja = (ka < D_) ? (signal + (size_t)mrow * D_ + ka)
                                    : (hidden + (size_t)mrow * D_ + ka - D_);
        const float* jc = (kc < D_) ? (signal + (size_t)mrow * D_ + kc)
                                    : (hidden + (size_t)mrow * D_ + kc - D_);
        float4 a0 = ((const float4*)ja)[0], a1 = ((const float4*)ja)[1];
        float4 c0 = ((const float4*)jc)[0], c1 = ((const float4*)jc)[1];
        // B[k, n] = W[n, k] (W row-major D x 2D), contiguous in k
        const float* pk0 = key_W + (size_t)ncol * (2*D_) + ka;
        const float* pk1 = key_W + (size_t)ncol * (2*D_) + kc;
        const float* pv0 = val_W + (size_t)ncol * (2*D_) + ka;
        const float* pv1 = val_W + (size_t)ncol * (2*D_) + kc;
        float4 k00 = ((const float4*)pk0)[0], k01 = ((const float4*)pk0)[1];
        float4 k10 = ((const float4*)pk1)[0], k11 = ((const float4*)pk1)[1];
        float4 v00 = ((const float4*)pv0)[0], v01 = ((const float4*)pv0)[1];
        float4 v10 = ((const float4*)pv1)[0], v11 = ((const float4*)pv1)[1];

        v16bf A  = pack16(a0, a1, c0, c1);
        v16bf Bk = pack16(k00, k01, k10, k11);
        v16bf Bv = pack16(v00, v01, v10, v11);
#if defined(__HIP_DEVICE_COMPILE__)
        acck = __builtin_amdgcn_wmma_f32_16x16x32_bf16(false, A, false, Bk, (short)0, acck, false, false);
        accv = __builtin_amdgcn_wmma_f32_16x16x32_bf16(false, A, false, Bv, (short)0, accv, false, false);
#endif
    }
    const float kbias = key_b[ncol];
    const float vbias = val_b[ncol];
    const int half = lane >> 4;
    const int d = ncol;
#pragma unroll
    for (int r = 0; r < 8; ++r) {
        const int b = tm * 16 + r + 8 * half;   // batch row of this element
        const float cg = scal[b*16 + 2];
        const int   si = __float_as_int(scal[b*16 + 0]);
        const float bkey = tanhf(acck[r] + kbias);
        const float bval = tanhf(accv[r] + vbias);
        const float sk = ep_keys[((size_t)b * M_ + si) * D_ + d];
        const float sv = ep_vals[((size_t)b * M_ + si) * D_ + d];
        cand_key[(size_t)b * D_ + d] = (1.0f - cg) * bkey + cg * sk;        // mixed (pre-norm)
        cand_val[(size_t)b * D_ + d] = tanhf((1.0f - 0.35f*cg) * bval + 0.35f*cg * sv);
    }
}

// ---------------- Kernel 4: similarity + cand inv-norm + target scalars ---------
__global__ __launch_bounds__(256) void k_target(
    const float* __restrict__ cand_key, const float* __restrict__ keys,
    const float* __restrict__ inv_norm, const float* __restrict__ s_strength,
    const float* __restrict__ s_age, const float* __restrict__ s_usage,
    const float* __restrict__ delay, float* __restrict__ scal) {
    const int b = blockIdx.x, t = threadIdx.x;
    const int wave = t >> 5, lane = t & 31;
    __shared__ float s_sim[N_], s_rep[N_];
    __shared__ float s_cinv;
    const float4* cp = (const float4*)(cand_key + (size_t)b * D_) + lane * 4;
    const float4 c0 = cp[0], c1 = cp[1], c2 = cp[2], c3 = cp[3];
    // wave 0: candidate key inverse norm (argmax below is scale-invariant,
    // cinv only matters for the 0.81 threshold and the final update kernel)
    if (wave == 0) {
        float ss = dot4(c0,c0) + dot4(c1,c1) + dot4(c2,c2) + dot4(c3,c3);
        ss = wave_sum(ss);
        if (lane == 0) {
            const float ci = 1.0f / fmaxf(sqrtf(ss), 1e-6f);
            s_cinv = ci;
            scal[b*16 + 6] = ci;
        }
    }
#pragma unroll
    for (int i = 0; i < 8; ++i) {
        const int n = wave + 8 * i;
        const float4* kp = (const float4*)(keys + ((size_t)b * N_ + n) * D_) + lane * 4;
        float d = dot4(c0, kp[0]) + dot4(c1, kp[1]) + dot4(c2, kp[2]) + dot4(c3, kp[3]);
        d = wave_sum(d);
        if (lane == 0) s_sim[n] = d * inv_norm[b*N_+n];
    }
    if (t < N_)
        s_rep[t] = 1.3f * s_age[b*N_+t] + (1.f - s_strength[b*N_+t]) + 0.9f * (1.f - s_usage[b*N_+t]);
    __syncthreads();
    if (t == 0) {
        int mi = 0; float ms = s_sim[0];
        for (int n = 1; n < N_; ++n) if (s_sim[n] > ms) { ms = s_sim[n]; mi = n; }
        int ri = 0; float rs = s_rep[0];
        for (int n = 1; n < N_; ++n) if (s_rep[n] > rs) { rs = s_rep[n]; ri = n; }
        const bool um = (ms * s_cinv) > 0.81f;
        const int  ti = um ? mi : ri;
        const float p = scal[b*16+3], kf = scal[b*16+4], comp = scal[b*16+5];
        const float ow = (0.1f + 0.8f*kf) * (0.55f + 0.45f*comp);
        scal[b*16+7]  = __int_as_float(ti);
        scal[b*16+8]  = ow;
        scal[b*16+9]  = um ? (0.18f + 0.24f*p) : (0.78f + 0.1f*p);   // key_mix
        scal[b*16+10] = um ? (0.34f + 0.22f*p) : (0.82f + 0.1f*p);   // value_mix
        scal[b*16+11] = ow * (0.55f + 0.2f*kf + 0.15f*p);            // strength boost
        scal[b*16+12] = ow * (0.6f + 0.4f*delay[b]);                 // usage add
    }
}

// ---------------- Kernel 5: final per-slot updates (float4, 2 rows / block) -----
__global__ __launch_bounds__(256) void k_update(
    const float* __restrict__ keys, const float* __restrict__ values,
    const float* __restrict__ s_strength, const float* __restrict__ s_age,
    const float* __restrict__ s_usage, const float* __restrict__ cand_key,
    const float* __restrict__ cand_val, const float* __restrict__ scal,
    float* __restrict__ out_keys, float* __restrict__ out_vals,
    float* __restrict__ out_str, float* __restrict__ out_age, float* __restrict__ out_usage) {
    const int lrow = threadIdx.x >> 7;            // 0 or 1
    const int tt   = threadIdx.x & 127;           // 128 x float4 = 512 floats
    const int id   = blockIdx.x * 2 + lrow;       // row id in [0, B*N)
    const int b = id >> 6, n = id & 63;
    __shared__ float s_par[2][3];                 // okm, ovm, cinv per row

    if (tt == 0) {
        const int ti = __float_as_int(scal[b*16 + 7]);
        const bool is_t = (n == ti);
        const float ow = is_t ? scal[b*16 + 8] : 0.0f;
        s_par[lrow][0] = ow * scal[b*16 + 9];
        s_par[lrow][1] = ow * scal[b*16 + 10];
        s_par[lrow][2] = scal[b*16 + 6];
        const float sbv = is_t ? scal[b*16 + 11] : 0.0f;
        const float uav = is_t ? scal[b*16 + 12] : 0.0f;
        out_str[id]   = clamp01(s_strength[id] * 0.97f + sbv);
        out_usage[id] = clamp01(s_usage[id] * 0.96f + uav);
        out_age[id]   = clamp01((s_age[id] + 0.02f) * (1.0f - 0.85f * ow));
    }
    __syncthreads();
    const float okm  = s_par[lrow][0];
    const float ovm  = s_par[lrow][1];
    const float cinv = s_par[lrow][2];

    const float4 k4  = ((const float4*)(keys     + (size_t)id * D_))[tt];
    const float4 ck4 = ((const float4*)(cand_key + (size_t)b  * D_))[tt];
    float4 rk;
    rk.x = k4.x + okm * (ck4.x * cinv - k4.x);
    rk.y = k4.y + okm * (ck4.y * cinv - k4.y);
    rk.z = k4.z + okm * (ck4.z * cinv - k4.z);
    rk.w = k4.w + okm * (ck4.w * cinv - k4.w);
    ((float4*)(out_keys + (size_t)id * D_))[tt] = rk;

    const float4 v4  = ((const float4*)(values   + (size_t)id * D_))[tt];
    const float4 cv4 = ((const float4*)(cand_val + (size_t)b  * D_))[tt];
    float4 rv;
    rv.x = v4.x + ovm * (cv4.x - v4.x);
    rv.y = v4.y + ovm * (cv4.y - v4.y);
    rv.z = v4.z + ovm * (cv4.z - v4.z);
    rv.w = v4.w + ovm * (cv4.w - v4.w);
    ((float4*)(out_vals + (size_t)id * D_))[tt] = rv;
}

extern "C" void kernel_launch(void* const* d_in, const int* in_sizes, int n_in,
                              void* d_out, int out_size, void* d_ws, size_t ws_size,
                              hipStream_t stream) {
    const float* query_key   = (const float*)d_in[0];
    const float* delay_gate  = (const float*)d_in[1];
    const float* sal_gate    = (const float*)d_in[2];
    const float* short_keys  = (const float*)d_in[3];
    const float* short_vals  = (const float*)d_in[4];
    const float* short_str   = (const float*)d_in[5];
    const float* short_age   = (const float*)d_in[6];
    const float* short_usage = (const float*)d_in[7];
    const float* signal      = (const float*)d_in[8];
    const float* hidden      = (const float*)d_in[9];
    const float* branch      = (const float*)d_in[10];
    const float* entropy     = (const float*)d_in[11];
    const float* ep_keys     = (const float*)d_in[12];
    const float* ep_vals     = (const float*)d_in[13];
    const float* ep_str      = (const float*)d_in[14];
    const float* ep_hits     = (const float*)d_in[15];
    const float* ep_age      = (const float*)d_in[16];
    const float* key_W       = (const float*)d_in[17];
    const float* key_b       = (const float*)d_in[18];
    const float* val_W       = (const float*)d_in[19];
    const float* val_b       = (const float*)d_in[20];
    const float* fg_W        = (const float*)d_in[21];
    const float* fg_b        = (const float*)d_in[22];
    const float* pg_W        = (const float*)d_in[23];
    const float* pg_b        = (const float*)d_in[24];
    const float* cg_W        = (const float*)d_in[25];
    const float* cg_b        = (const float*)d_in[26];

    float* ws       = (float*)d_ws;
    float* qdot     = ws;                         // B*N
    float* inv_norm = qdot + B_ * N_;             // B*N
    float* scal     = inv_norm + B_ * N_;         // B*16
    float* cand_key = scal + 16 * B_;             // B*D
    float* cand_val = cand_key + B_ * D_;         // B*D

    float* out      = (float*)d_out;
    float* out_read = out;
    float* out_hits = out_read + (size_t)B_ * D_;
    float* out_keys = out_hits + (size_t)B_ * N_;
    float* out_vals = out_keys + (size_t)B_ * N_ * D_;
    float* out_str  = out_vals + (size_t)B_ * N_ * D_;
    float* out_age  = out_str  + (size_t)B_ * N_;
    float* out_usg  = out_age  + (size_t)B_ * N_;

    k_dotnorm<<<(B_ * N_) / 8, 256, 0, stream>>>(query_key, short_keys, qdot, inv_norm);

    k_perbatch<<<B_, 256, 0, stream>>>(qdot, inv_norm, delay_gate, sal_gate,
        short_str, short_age, short_usage, short_vals,
        signal, hidden, branch, entropy,
        ep_vals, ep_str, ep_hits, ep_age,
        fg_W, fg_b, pg_W, pg_b, cg_W, cg_b,
        scal, out_read, out_hits);

    // 128 M-tiles * 32 N-tiles = 4096 waves, 8 waves per block
    k_gemm_mix<<<(128 * 32) / 8, 256, 0, stream>>>(signal, hidden,
        key_W, key_b, val_W, val_b, ep_keys, ep_vals, scal, cand_key, cand_val);

    k_target<<<B_, 256, 0, stream>>>(cand_key, short_keys, inv_norm,
        short_str, short_age, short_usage, delay_gate, scal);

    k_update<<<(B_ * N_) / 2, 256, 0, stream>>>(short_keys, short_vals,
        short_str, short_age, short_usage, cand_key, cand_val, scal,
        out_keys, out_vals, out_str, out_age, out_usg);

    (void)in_sizes; (void)n_in; (void)out_size; (void)ws_size;
}
